// SelfAttention_34333968564399
// MI455X (gfx1250) — matrix operand order repaired and checked by reading, not compiled, not run on hardware
//
#include <hip/hip_runtime.h>
#include <hip/hip_bf16.h>

// ---------------- problem constants ----------------
#define BB    2
#define SS    2048
#define HH    16
#define DQ    64
#define DM    1024
#define MTOT  (BB*SS)                 // 4096 rows of x
#define A_ELEMS ((size_t)BB*HH*SS*SS) // 134,217,728 attention probs

typedef __attribute__((ext_vector_type(16))) __bf16 v16bf_t;
typedef __attribute__((ext_vector_type(8)))  float  v8f_t;
typedef __attribute__((ext_vector_type(4)))  unsigned int su32x4;
typedef __attribute__((ext_vector_type(8)))  unsigned int su32x8;

union Frag { v16bf_t v; unsigned int u[8]; };

static __device__ __forceinline__ unsigned short f2bf(float f) {
  union { float f; unsigned u; } c; c.f = f;
  unsigned r = c.u + 0x7FFFu + ((c.u >> 16) & 1u);   // round-to-nearest-even
  return (unsigned short)(r >> 16);
}
static __device__ __forceinline__ unsigned pack2bf(float lo, float hi) {
  return (unsigned)f2bf(lo) | ((unsigned)f2bf(hi) << 16);
}
static __device__ __forceinline__ v8f_t wmma_bf16(const Frag& a, const Frag& b, v8f_t c) {
  // D = A(16x32 bf16) x B(32x16 bf16) + C(16x16 f32)
  return __builtin_amdgcn_wmma_f32_16x16x32_bf16(false, a.v, false, b.v,
                                                 (short)0, c, false, false);
}

// ---- Tensor Data Mover: 2D tile (tile_d0 x tile_d1 elements, 2B/elt) -> LDS ----
// D# per CDNA5 ISA ch.8: group0 = {flags/lds/global/type}, group1 = dims/strides.
static __device__ __forceinline__ void tdm_load_2d_bf16(unsigned lds_off,
                                                        const void* gaddr,
                                                        unsigned tensor_d0, unsigned tensor_d1,
                                                        unsigned stride_d0,
                                                        unsigned tile_d0, unsigned tile_d1) {
  unsigned long long ga = (unsigned long long)(uintptr_t)gaddr;
  su32x4 g0;
  g0.x = 1u;                                               // count=1, user desc, no gather
  g0.y = lds_off;                                          // lds_addr (bytes)
  g0.z = (unsigned)ga;                                     // global_addr[31:0]
  g0.w = (unsigned)((ga >> 32) & 0x01FFFFFFu) | (2u << 30); // global_addr[56:32] | type=2
  su32x8 g1;
  g1.s0 = (1u << 16);                                      // data_size = 2 bytes
  g1.s1 = (tensor_d0 & 0xFFFFu) << 16;                     // tensor_dim0[15:0]
  g1.s2 = (tensor_d0 >> 16) | ((tensor_d1 & 0xFFFFu) << 16); // td0 hi | td1 lo
  g1.s3 = (tensor_d1 >> 16) | (tile_d0 << 16);             // td1 hi | tile_dim0
  g1.s4 = tile_d1;                                         // tile_dim1 (tile_dim2=0)
  g1.s5 = stride_d0;                                       // tensor_dim0_stride[31:0]
  g1.s6 = 0u;                                              // stride hi, dim1_stride lo
  g1.s7 = 0u;
  asm volatile("tensor_load_to_lds %0, %1" :: "s"(g0), "s"(g1) : "memory");
}

// ---------------- fp32 -> bf16 convert ----------------
__global__ __launch_bounds__(256) void k_conv(const float* __restrict__ src,
                                              unsigned short* __restrict__ dst, int n) {
  int i = blockIdx.x * blockDim.x + threadIdx.x;
  int stride = gridDim.x * blockDim.x;
  for (; i < n; i += stride) dst[i] = f2bf(src[i]);
}

// ---------------- GEMM: C[M=4096,N=1024] = A[M,1024] * W[N,1024]^T ----------------
// A/W tiles staged into LDS by the Tensor Data Mover (one wave issues, all consume).
// mode 0: write bf16 head-major  out[((b*16+h)*2048+s)*64+d]      (Q, K)
// mode 1: write bf16 d-major     out[((b*16+h)*64+d)*2048+s]      (V transposed)
// mode 2: write fp32 row-major   out[M*1024+N]                    (fc)
__global__ __launch_bounds__(256) void k_gemm(const unsigned short* __restrict__ Abf,
                                              const unsigned short* __restrict__ Wbf,
                                              void* __restrict__ out, int mode, int mRows) {
  __shared__ __align__(16) unsigned short As[64 * 32];
  __shared__ __align__(16) unsigned short Ws[64 * 32];

  const int t = threadIdx.x;
  const int lane = t & 31, w = t >> 5;
  const int wm = w & 1, wn = w >> 1;                 // 2x4 wave grid, wave tile 32x16
  const int lrow = lane & 15;
  const int kb = (lane >> 4) * 8;
  const int khalf = (lane >> 4) * 16;
  const int mBase = blockIdx.x * 64;
  const int nBase = blockIdx.y * 64;

  const unsigned ldsA = (unsigned)(uintptr_t)As;      // generic->LDS: low 32 bits
  const unsigned ldsW = (unsigned)(uintptr_t)Ws;

  v8f_t z = {0.f,0.f,0.f,0.f,0.f,0.f,0.f,0.f};
  v8f_t acc0 = z, acc1 = z;

  for (int kpos = 0; kpos < DM; kpos += 32) {
    if (w == 0) {        // wave 0 drives the TDM; descriptors are wave-uniform SGPRs
      tdm_load_2d_bf16(ldsA, Abf + (size_t)mBase * DM + kpos, DM, (unsigned)mRows, DM, 32, 64);
      tdm_load_2d_bf16(ldsW, Wbf + (size_t)nBase * DM + kpos, DM, DM,              DM, 32, 64);
      __builtin_amdgcn_s_wait_tensorcnt(0);
    }
    __syncthreads();     // publish LDS tiles to all waves

    Frag a0, a1, bf;
    const unsigned short* ap0 = &As[(wm * 32 + 0  + lrow) * 32];
    const unsigned short* ap1 = &As[(wm * 32 + 16 + lrow) * 32];
    const unsigned short* bp  = &Ws[(wn * 16 + lrow) * 32 + kb];
#pragma unroll
    for (int j = 0; j < 4; ++j) {
      a0.u[j]     = *(const unsigned*)(ap0 + kb + 2 * j);
      a0.u[4 + j] = *(const unsigned*)(ap0 + kb + 16 + 2 * j);
      a1.u[j]     = *(const unsigned*)(ap1 + kb + 2 * j);
      a1.u[4 + j] = *(const unsigned*)(ap1 + kb + 16 + 2 * j);
    }
#pragma unroll
    for (int j = 0; j < 8; ++j) bf.u[j] = *(const unsigned*)(bp + 2 * (j & 3) + (j >> 2) * 16);

    acc0 = wmma_bf16(a0, bf, acc0);
    acc1 = wmma_bf16(a1, bf, acc1);
    __syncthreads();     // all reads done before next TDM overwrites the tiles
  }

  // epilogue
#pragma unroll
  for (int sub = 0; sub < 2; ++sub) {
    v8f_t acc = sub ? acc1 : acc0;
#pragma unroll
    for (int r = 0; r < 8; ++r) {
      int M = mBase + wm * 32 + sub * 16 + r + 8 * (lane >> 4);
      int N = nBase + wn * 16 + lrow;
      float val = acc[r];
      if (mode == 2) {
        ((float*)out)[(size_t)M * DM + N] = val;
      } else {
        int b = M >> 11, s = M & 2047, h = N >> 6, d = N & 63;
        size_t idx = (mode == 0)
          ? (((size_t)(b * HH + h) * SS + s) * DQ + d)     // head-major
          : (((size_t)(b * HH + h) * DQ + d) * SS + s);    // d-major (V^T)
        ((unsigned short*)out)[idx] = f2bf(val);
      }
    }
  }
}

// ---------------- fused attention: scores -> softmax -> a-store -> P*V ----------------
// One workgroup owns 16 query rows of one (b,h). The full 16x2048 fp32 score
// strip lives in LDS (128 KB; CDNA5 WGP has 320 KB), so `a` is written exactly
// once (normalized) and P feeds WMMA straight from LDS.
__global__ __launch_bounds__(256) void k_attn(const unsigned short* __restrict__ Qh,
                                              const unsigned short* __restrict__ Kh,
                                              const unsigned short* __restrict__ Vt,
                                              float* __restrict__ a,
                                              unsigned short* __restrict__ Obf) {
  __shared__ __align__(16) float sc[16 * 2048];   // 128 KB score/prob strip
  const int bh = blockIdx.y;
  const int b = bh >> 4, h = bh & 15;
  const int row0 = blockIdx.x * 16;

  const int t = threadIdx.x;
  const int lane = t & 31, w = t >> 5;
  const int lrow = lane & 15;
  const int kb = (lane >> 4) * 8;
  const int khalf = (lane >> 4) * 16;

  const unsigned short* Qb = Qh + (size_t)bh * SS * DQ;
  const unsigned short* Kb = Kh + (size_t)bh * SS * DQ;

  // Q A-fragments (16 rows x 64 d), resident for the whole kernel
  Frag qa0, qa1;
  {
    const unsigned short* qp = Qb + (size_t)(row0 + lrow) * DQ;
#pragma unroll
    for (int j = 0; j < 4; ++j) {
      qa0.u[j]     = *(const unsigned*)(qp + kb + 2 * j);
      qa0.u[4 + j] = *(const unsigned*)(qp + kb + 16 + 2 * j);
      qa1.u[j]     = *(const unsigned*)(qp + 32 + kb + 2 * j);
      qa1.u[4 + j] = *(const unsigned*)(qp + 32 + kb + 16 + 2 * j);
    }
  }

  // ---- phase 1: scores = Q K^T, wave w covers columns [w*256, w*256+256) ----
  for (int nt = 0; nt < 16; ++nt) {
    const int col = w * 256 + nt * 16;
    Frag kf0, kf1;
    const unsigned short* kp = Kb + (size_t)(col + lrow) * DQ;
#pragma unroll
    for (int j = 0; j < 8; ++j) {
      kf0.u[j] = *(const unsigned*)(kp + kb + 2 * (j & 3) + (j >> 2) * 16);
      kf1.u[j] = *(const unsigned*)(kp + 32 + kb + 2 * (j & 3) + (j >> 2) * 16);
    }
    v8f_t acc = {0.f,0.f,0.f,0.f,0.f,0.f,0.f,0.f};
    acc = wmma_bf16(qa0, kf0, acc);
    acc = wmma_bf16(qa1, kf1, acc);
#pragma unroll
    for (int r = 0; r < 8; ++r)
      sc[(r + 8 * (lane >> 4)) * 2048 + col + lrow] = acc[r];
  }
  __syncthreads();

  // ---- phase 2: softmax over full rows, in LDS ----
  {
    const int row = t >> 4, j = t & 15;
    float mx = -3.0e38f;
    for (int c = j; c < 2048; c += 16) mx = fmaxf(mx, sc[row * 2048 + c]);
#pragma unroll
    for (int m = 1; m < 16; m <<= 1) mx = fmaxf(mx, __shfl_xor(mx, m, 16));
    float ssum = 0.f;
    for (int c = j; c < 2048; c += 16) ssum += __expf(sc[row * 2048 + c] - mx);
#pragma unroll
    for (int m = 1; m < 16; m <<= 1) ssum += __shfl_xor(ssum, m, 16);
    const float invl = 1.0f / ssum;
    for (int c = j; c < 2048; c += 16)
      sc[row * 2048 + c] = __expf(sc[row * 2048 + c] - mx) * invl;
  }
  __syncthreads();

  // ---- coalesced single store of normalized probabilities ----
  {
    float* abase = a + (size_t)(bh * SS + row0) * SS;   // LDS stride == global stride
#pragma unroll 4
    for (int i4 = t; i4 < 16 * 2048 / 4; i4 += 256)
      ((float4*)abase)[i4] = ((const float4*)sc)[i4];
  }

  // ---- phase 3: O = P * V, wave w accumulates its own column range ----
  v8f_t z = {0.f,0.f,0.f,0.f,0.f,0.f,0.f,0.f};
  v8f_t acc[4] = {z, z, z, z};
  for (int ch = 0; ch < 8; ++ch) {
    const int c0 = w * 256 + ch * 32;
    Frag pf;
    const float* prow = &sc[lrow * 2048 + c0];
#pragma unroll
    for (int j = 0; j < 4; ++j) {
      pf.u[j]     = pack2bf(prow[kb + 2 * j],      prow[kb + 2 * j + 1]);
      pf.u[4 + j] = pack2bf(prow[16 + kb + 2 * j], prow[16 + kb + 2 * j + 1]);
    }
#pragma unroll
    for (int nt = 0; nt < 4; ++nt) {
      Frag vf;
      const unsigned short* vp = Vt + (size_t)(bh * DQ + nt * 16 + lrow) * SS + c0 + kb;
#pragma unroll
      for (int j = 0; j < 8; ++j) vf.u[j] = *(const unsigned*)(vp + 2 * (j & 3) + (j >> 2) * 16);
      acc[nt] = wmma_bf16(pf, vf, acc[nt]);
    }
  }
  __syncthreads();                      // done reading sc; reuse it as reduce buffer

  float* red = sc;                      // 8*16*64 = 8192 floats (32 KB) reused
#pragma unroll
  for (int nt = 0; nt < 4; ++nt)
#pragma unroll
    for (int r = 0; r < 8; ++r)
      red[(w * 16 + r + 8 * (lane >> 4)) * 64 + nt * 16 + lrow] = acc[nt][r];
  __syncthreads();

  {
    const int row = t >> 4, d0 = (t & 15) * 4;
#pragma unroll
    for (int dd = 0; dd < 4; ++dd) {
      int d = d0 + dd;
      float sum = 0.f;
#pragma unroll
      for (int ww = 0; ww < 8; ++ww) sum += red[(ww * 16 + row) * 64 + d];
      Obf[(size_t)(b * SS + row0 + row) * DM + h * DQ + d] = f2bf(sum);
    }
  }
}

// ---------------- residual + LayerNorm ----------------
__global__ __launch_bounds__(256) void k_ln(const float* __restrict__ Y,
                                            const float* __restrict__ xq,
                                            float* __restrict__ out) {
  __shared__ float rbuf[256];
  const int row = blockIdx.x;
  const int t = threadIdx.x;
  float vals[4];
  float s = 0.f;
#pragma unroll
  for (int i = 0; i < 4; ++i) {
    int c = t + i * 256;
    float v = Y[(size_t)row * DM + c] + xq[(size_t)row * DM + c];
    vals[i] = v; s += v;
  }
  rbuf[t] = s; __syncthreads();
  for (int ofs = 128; ofs > 0; ofs >>= 1) { if (t < ofs) rbuf[t] += rbuf[t + ofs]; __syncthreads(); }
  const float mu = rbuf[0] * (1.0f / DM);
  float vs = 0.f;
#pragma unroll
  for (int i = 0; i < 4; ++i) { float d = vals[i] - mu; vs += d * d; }
  __syncthreads();
  rbuf[t] = vs; __syncthreads();
  for (int ofs = 128; ofs > 0; ofs >>= 1) { if (t < ofs) rbuf[t] += rbuf[t + ofs]; __syncthreads(); }
  const float rstd = rsqrtf(rbuf[0] * (1.0f / DM) + 1e-5f);
#pragma unroll
  for (int i = 0; i < 4; ++i) {
    int c = t + i * 256;
    out[(size_t)row * DM + c] = (vals[i] - mu) * rstd;
  }
}

// ---------------- launch ----------------
extern "C" void kernel_launch(void* const* d_in, const int* in_sizes, int n_in,
                              void* d_out, int out_size, void* d_ws, size_t ws_size,
                              hipStream_t stream) {
  const float* xq  = (const float*)d_in[0];
  const float* xk  = (const float*)d_in[1];
  const float* xv  = (const float*)d_in[2];
  const float* W_Q = (const float*)d_in[3];
  const float* W_K = (const float*)d_in[4];
  const float* W_V = (const float*)d_in[5];
  const float* Wfc = (const float*)d_in[6];

  float* a_out = (float*)d_out;                 // [B,H,S,S]
  float* final_out = a_out + A_ELEMS;           // [B,S,DM]

  unsigned char* ws = (unsigned char*)d_ws;
  size_t off = 0;
  auto alloc = [&](size_t bytes) { size_t o = off; off = (off + bytes + 255) & ~(size_t)255; return o; };

  const size_t XBF = (size_t)MTOT * DM * 2, WBF = (size_t)DM * DM * 2;
  unsigned short* xq_bf = (unsigned short*)(ws + alloc(XBF));
  unsigned short* xk_bf = (unsigned short*)(ws + alloc(XBF));
  unsigned short* xv_bf = (unsigned short*)(ws + alloc(XBF));
  unsigned short* wq_bf = (unsigned short*)(ws + alloc(WBF));
  unsigned short* wk_bf = (unsigned short*)(ws + alloc(WBF));
  unsigned short* wv_bf = (unsigned short*)(ws + alloc(WBF));
  unsigned short* wf_bf = (unsigned short*)(ws + alloc(WBF));
  unsigned short* Qh    = (unsigned short*)(ws + alloc((size_t)BB*HH*SS*DQ*2));
  unsigned short* Kh    = (unsigned short*)(ws + alloc((size_t)BB*HH*SS*DQ*2));
  unsigned short* Vt    = (unsigned short*)(ws + alloc((size_t)BB*HH*SS*DQ*2));
  unsigned short* Obf   = (unsigned short*)(ws + alloc(XBF));
  float*          Yf    = (float*)(ws + alloc((size_t)MTOT*DM*4));

  // fp32 -> bf16
  k_conv<<<1024, 256, 0, stream>>>(xq,  xq_bf, MTOT * DM);
  k_conv<<<1024, 256, 0, stream>>>(xk,  xk_bf, MTOT * DM);
  k_conv<<<1024, 256, 0, stream>>>(xv,  xv_bf, MTOT * DM);
  k_conv<<<512,  256, 0, stream>>>(W_Q, wq_bf, DM * DM);
  k_conv<<<512,  256, 0, stream>>>(W_K, wk_bf, DM * DM);
  k_conv<<<512,  256, 0, stream>>>(W_V, wv_bf, DM * DM);
  k_conv<<<512,  256, 0, stream>>>(Wfc, wf_bf, DM * DM);

  // projections (WMMA GEMMs, TDM-staged tiles)
  dim3 gg(MTOT / 64, DM / 64);
  k_gemm<<<gg, 256, 0, stream>>>(xq_bf, wq_bf, (void*)Qh, 0, MTOT);
  k_gemm<<<gg, 256, 0, stream>>>(xk_bf, wk_bf, (void*)Kh, 0, MTOT);
  k_gemm<<<gg, 256, 0, stream>>>(xv_bf, wv_bf, (void*)Vt, 1, MTOT);

  // fused attention (scores + softmax + a store + P*V)
  dim3 ga(SS / 16, BB * HH);
  k_attn<<<ga, 256, 0, stream>>>(Qh, Kh, Vt, a_out, Obf);

  // output projection + residual layernorm
  k_gemm<<<gg, 256, 0, stream>>>(Obf, wf_bf, (void*)Yf, 2, MTOT);
  k_ln<<<MTOT, 256, 0, stream>>>(Yf, xq, final_out);
}
